// GAT_52020643889237
// MI455X (gfx1250) — compile-verified, matrix operand order checked by reading, not run on hardware
//
#include <hip/hip_runtime.h>

#define NN      4096
#define FIN     512
#define HIDD    64
#define NHEADS  8
#define LRALPHA 0.2f
#define NEG_BIG (-3.0e38f)

typedef __attribute__((ext_vector_type(16))) _Float16 v16h;
typedef __attribute__((ext_vector_type(8)))  _Float16 h8;
typedef __attribute__((ext_vector_type(8)))  float    v8f;
typedef __attribute__((ext_vector_type(4)))  float    f4;

// ---------------- Kernel 0a: pack adjacency into bitmask (read 67MB once) ---
__global__ __launch_bounds__(256) void gat_pack_adj(const float* __restrict__ adj,
                                                    unsigned* __restrict__ bits) {
    int gtid = blockIdx.x * 256 + threadIdx.x;
    int lane = gtid & 31;
    int w    = gtid >> 5;                 // word index, one word per wave32
    int i    = w >> 7;                    // row (NN/32 = 128 words per row)
    int j0   = (w & 127) << 5;
    float v  = adj[(size_t)i * NN + j0 + lane];
    unsigned long long m = __ballot(v > 0.0f);
    if (lane == 0) bits[w] = (unsigned)m;
}

// ---------------- Kernel 0b: W [H,FIN,HID] f32 -> Wt [H*HID, FIN] f16 -------
__global__ __launch_bounds__(256) void gat_transpose_w(const float* __restrict__ W,
                                                       _Float16* __restrict__ Wt) {
    int t  = blockIdx.x * 256 + threadIdx.x;   // 0 .. 8*64*512-1
    int k  = t & (FIN - 1);
    int hd = t >> 9;                           // h*64+d
    int h  = hd >> 6;
    int d  = hd & 63;
    Wt[t] = (_Float16)W[((size_t)h * FIN + k) * HIDD + d];
}

// ---------------- Kernel 1: projection Wh = x@W (per head), + e_src/e_dst ---
// One wave per (head, 16-row block). Also stores WhT [H*HID, NN] in f16.
__global__ __launch_bounds__(32) void gat_project(const float* __restrict__ x,
                                                  const _Float16* __restrict__ Wt,
                                                  const float* __restrict__ avec,
                                                  _Float16* __restrict__ WhT,
                                                  float* __restrict__ e_src,
                                                  float* __restrict__ e_dst,
                                                  float* __restrict__ e_dstT) {
    const int lane = threadIdx.x;
    const int g    = lane >> 4;
    const int nlo  = lane & 15;
    const int h    = blockIdx.x >> 8;     // NN/16 = 256 row blocks
    const int row0 = (blockIdx.x & 255) * 16;

    v8f c[4];
#pragma unroll
    for (int db = 0; db < 4; ++db)
#pragma unroll
        for (int r = 0; r < 8; ++r) c[db][r] = 0.0f;

    const float* xrow = x + (size_t)(row0 + nlo) * FIN;

    for (int kk = 0; kk < FIN; kk += 32) {
        // A fragment: rows of x (m = lane&15), fp32 -> fp16
        f4 xa0 = *(const f4*)(xrow + kk + g * 8);
        f4 xa1 = *(const f4*)(xrow + kk + g * 8 + 4);
        f4 xb0 = *(const f4*)(xrow + kk + 16 + g * 8);
        f4 xb1 = *(const f4*)(xrow + kk + 16 + g * 8 + 4);
        v16h a;
#pragma unroll
        for (int i = 0; i < 4; ++i) {
            a[i]      = (_Float16)xa0[i];
            a[4 + i]  = (_Float16)xa1[i];
            a[8 + i]  = (_Float16)xb0[i];
            a[12 + i] = (_Float16)xb1[i];
        }
#pragma unroll
        for (int db = 0; db < 4; ++db) {
            const _Float16* wrow = Wt + ((size_t)(h * HIDD + db * 16 + nlo) * FIN) + kk + g * 8;
            h8 b0 = *(const h8*)(wrow);
            h8 b1 = *(const h8*)(wrow + 16);
            v16h b;
#pragma unroll
            for (int i = 0; i < 8; ++i) { b[i] = b0[i]; b[8 + i] = b1[i]; }
            c[db] = __builtin_amdgcn_wmma_f32_16x16x32_f16(
                false, a, false, b, (short)0, c[db], false, false);
        }
    }

    // Store WhT (f16, [h*64+d][node]) for the aggregation GEMM's B operand.
#pragma unroll
    for (int db = 0; db < 4; ++db) {
        int d = db * 16 + nlo;
#pragma unroll
        for (int r = 0; r < 8; ++r)
            WhT[(size_t)(h * HIDD + d) * NN + row0 + r + 8 * g] = (_Float16)c[db][r];
    }

    // e_src / e_dst: dot(Wh_row, a1/a2) — 16-lane reduction per row.
    float a1[4], a2[4];
#pragma unroll
    for (int db = 0; db < 4; ++db) {
        a1[db] = avec[h * 128 + db * 16 + nlo];
        a2[db] = avec[h * 128 + 64 + db * 16 + nlo];
    }
#pragma unroll
    for (int r = 0; r < 8; ++r) {
        float s1 = 0.0f, s2 = 0.0f;
#pragma unroll
        for (int db = 0; db < 4; ++db) {
            s1 += c[db][r] * a1[db];
            s2 += c[db][r] * a2[db];
        }
#pragma unroll
        for (int msk = 1; msk <= 8; msk <<= 1) {
            s1 += __shfl_xor(s1, msk, 32);
            s2 += __shfl_xor(s2, msk, 32);
        }
        if (nlo == 0) {
            int node = row0 + r + 8 * g;
            e_src[h * NN + node]  = s1;
            e_dst[h * NN + node]  = s2;
            e_dstT[node * NHEADS + h] = s2;   // [node][head] for the row-max pass
        }
    }
}

// ---------------- Kernel 1b: exact per-row masked max of e_dst (all heads) --
// m_i = max_{j in N(i)} e_dst[h][j]; one wave per row, ~41 set bits per row.
__global__ __launch_bounds__(256) void gat_rowmax(const unsigned* __restrict__ adjBits,
                                                  const float* __restrict__ e_dstT,
                                                  float* __restrict__ mN) {
    const int lane = threadIdx.x & 31;
    const int row  = blockIdx.x * 8 + (threadIdx.x >> 5);

    float m8[8];
#pragma unroll
    for (int h = 0; h < 8; ++h) m8[h] = NEG_BIG;

    for (int w = lane; w < NN / 32; w += 32) {
        unsigned bits = adjBits[row * (NN / 32) + w];
        while (bits) {
            int b = __builtin_ctz(bits);
            bits &= bits - 1;
            int j = (w << 5) + b;
            const f4* p = (const f4*)(e_dstT + (size_t)j * NHEADS);
            f4 v0 = p[0], v1 = p[1];
#pragma unroll
            for (int h = 0; h < 4; ++h) {
                m8[h]     = fmaxf(m8[h], v0[h]);
                m8[4 + h] = fmaxf(m8[4 + h], v1[h]);
            }
        }
    }
#pragma unroll
    for (int msk = 1; msk <= 16; msk <<= 1)
#pragma unroll
        for (int h = 0; h < 8; ++h) m8[h] = fmaxf(m8[h], __shfl_xor(m8[h], msk, 32));
    if (lane == 0) {
#pragma unroll
        for (int h = 0; h < 8; ++h) mN[h * NN + row] = m8[h];
    }
}

// ---------------- Kernel 2: one-pass masked softmax (exact max) + att@Wh ----
// One wave per (head, 16-row block); loops over 128 column tiles of 32.
__global__ __launch_bounds__(32) void gat_attend(const unsigned* __restrict__ adjBits,
                                                 const float* __restrict__ e_src,
                                                 const float* __restrict__ e_dst,
                                                 const float* __restrict__ mN,
                                                 const _Float16* __restrict__ WhT,
                                                 float* __restrict__ out) {
    const int lane = threadIdx.x;
    const int g    = lane >> 4;
    const int nlo  = lane & 15;
    const int h    = blockIdx.x >> 8;
    const int row0 = (blockIdx.x & 255) * 16;

    const float es = e_src[h * NN + row0 + nlo];
    // exact row max: lrelu is monotone => m_i = lrelu(es + max_neigh(e_dst))
    const float xm = es + mN[h * NN + row0 + nlo];
    const float mi = fmaxf(xm, LRALPHA * xm);
    const float esb = es - mi;           // u   = esb + ed
    const float bco = -(1.0f - LRALPHA) * mi;  // t-mi = max(u, LRALPHA*u + bco)

    const unsigned adjBase = (unsigned)(row0 + nlo) * (NN / 32);
    const float* edh = e_dst + h * NN;
    const _Float16* whh = WhT + (size_t)h * HIDD * NN;

    v8f c[4];
#pragma unroll
    for (int db = 0; db < 4; ++db)
#pragma unroll
        for (int r = 0; r < 8; ++r) c[db][r] = 0.0f;
    float li = 0.0f;

    for (int jt = 0; jt < NN / 32; ++jt) {
        const unsigned adjW = adjBits[adjBase + jt];
        if (__ballot(adjW != 0u) == 0ull) continue;   // empty 16x32 tile

        const int j0 = jt * 32;
        // e_dst values land directly in WMMA-A element order: two runs of 8
        f4 e0 = *(const f4*)(edh + j0 + g * 8);
        f4 e1 = *(const f4*)(edh + j0 + g * 8 + 4);
        f4 e2 = *(const f4*)(edh + j0 + 16 + g * 8);
        f4 e3 = *(const f4*)(edh + j0 + 16 + g * 8 + 4);
        float ed[16];
#pragma unroll
        for (int i = 0; i < 4; ++i) {
            ed[i] = e0[i]; ed[4 + i] = e1[i]; ed[8 + i] = e2[i]; ed[12 + i] = e3[i];
        }

        float rs = 0.0f;
        v16h aP;
#pragma unroll
        for (int e = 0; e < 16; ++e) {
            int k = g * 8 + (e & 7) + ((e >> 3) << 4);
            float u = esb + ed[e];
            float t = fmaxf(u, __builtin_fmaf(LRALPHA, u, bco));  // lrelu(x)-mi
            float p = ((adjW >> k) & 1u) ? __expf(t) : 0.0f;
            rs += p;
            aP[e] = (_Float16)p;
        }
        rs += __shfl_xor(rs, 16, 32);
        li += rs;

#pragma unroll
        for (int db = 0; db < 4; ++db) {
            const _Float16* wr = whh + (size_t)(db * 16 + nlo) * NN + j0 + g * 8;
            h8 b0 = *(const h8*)(wr);
            h8 b1 = *(const h8*)(wr + 16);
            v16h b;
#pragma unroll
            for (int i = 0; i < 8; ++i) { b[i] = b0[i]; b[8 + i] = b1[i]; }
            c[db] = __builtin_amdgcn_wmma_f32_16x16x32_f16(
                false, aP, false, b, (short)0, c[db], false, false);
        }
    }

    // epilogue: normalize by softmax denominator, ELU, write [n][h*64+d]
    float linv[8];
#pragma unroll
    for (int r = 0; r < 8; ++r) linv[r] = 1.0f / __shfl(li, r + 8 * g, 32);
#pragma unroll
    for (int db = 0; db < 4; ++db)
#pragma unroll
        for (int r = 0; r < 8; ++r) {
            float v = c[db][r] * linv[r];
            v = v > 0.0f ? v : (__expf(v) - 1.0f);
            out[(size_t)(row0 + r + 8 * g) * (NHEADS * HIDD) + h * HIDD + db * 16 + nlo] = v;
        }
}

extern "C" void kernel_launch(void* const* d_in, const int* in_sizes, int n_in,
                              void* d_out, int out_size, void* d_ws, size_t ws_size,
                              hipStream_t stream) {
    const float* x    = (const float*)d_in[0];   // [4096, 512]
    const float* adj  = (const float*)d_in[1];   // [4096, 4096]
    const float* W    = (const float*)d_in[2];   // [8, 512, 64]
    const float* avec = (const float*)d_in[3];   // [8, 128]
    float* out = (float*)d_out;                  // [4096, 512]

    char* ws = (char*)d_ws;
    size_t off = 0;
    _Float16* Wt     = (_Float16*)(ws + off); off += (size_t)NHEADS * HIDD * FIN * 2;  // 512 KB
    _Float16* WhT    = (_Float16*)(ws + off); off += (size_t)NHEADS * HIDD * NN * 2;   // 4 MB
    float*    e_src  = (float*)(ws + off);    off += (size_t)NHEADS * NN * 4;
    float*    e_dst  = (float*)(ws + off);    off += (size_t)NHEADS * NN * 4;
    float*    e_dstT = (float*)(ws + off);    off += (size_t)NN * NHEADS * 4;
    float*    mN     = (float*)(ws + off);    off += (size_t)NHEADS * NN * 4;
    unsigned* adjBit = (unsigned*)(ws + off);                                          // 2 MB

    // 0a: pack adjacency bitmask: 4096*128 words, one per wave32
    gat_pack_adj<<<(NN * (NN / 32)) / 8, 256, 0, stream>>>(adj, adjBit);
    // 0b: transpose W to f16 [h*64+d][k]
    gat_transpose_w<<<(NHEADS * HIDD * FIN) / 256, 256, 0, stream>>>(W, Wt);
    // 1: projection + attention logit vectors (one wave per head x 16-row block)
    gat_project<<<NHEADS * (NN / 16), 32, 0, stream>>>(x, Wt, avec, WhT, e_src, e_dst, e_dstT);
    // 1b: exact masked row-max of e_dst for all heads (one wave per row)
    gat_rowmax<<<NN / 8, 256, 0, stream>>>(adjBit, e_dstT, mN);
    // 2: one-pass masked softmax attention + aggregation + ELU
    gat_attend<<<NHEADS * (NN / 16), 32, 0, stream>>>(adjBit, e_src, e_dst, mN, WhT, out);
}
